// VolumeRenderer_56831007261366
// MI455X (gfx1250) — matrix operand order repaired and checked by reading, not compiled, not run on hardware
//
#include <hip/hip_runtime.h>

typedef _Float16 half_t;
typedef __attribute__((ext_vector_type(16))) _Float16 v16h;
typedef __attribute__((ext_vector_type(8)))  _Float16 v8h;
typedef __attribute__((ext_vector_type(8)))  float    v8f;

#define GLOBAL_AS __attribute__((address_space(1)))
#define LDS_AS    __attribute__((address_space(3)))
typedef const GLOBAL_AS _Float16 *gch_p;
typedef const GLOBAL_AS v16h     *gcv16h_p;
typedef const GLOBAL_AS float    *gcf_p;
typedef GLOBAL_AS float          *gf_p;
typedef const LDS_AS _Float16    *lch_p;
typedef LDS_AS _Float16          *lh_p;
typedef const LDS_AS v8h         *lcv8h_p;

#define M_RAYS   1024
#define NC       64
#define NF       192
#define ROWS     128     // points per block (8 M-tiles of 16)
#define NTHREADS 256     // 8 wave32s

// ---------------------------------------------------------------------------
// Weight repack: fp32 [K, Nsrc] row-major  ->  fp16 wT [Np][Kp]  (B^T panels)
// mode 1 = skip-layer remap: padded col 63 is the zero pad between pe and h.
// ---------------------------------------------------------------------------
__global__ void convert_w_kernel(const float* __restrict__ w, half_t* __restrict__ wT,
                                 int K, int Nsrc, int Kp, int Np, int mode)
{
    int i = blockIdx.x * 256 + threadIdx.x;
    if (i >= Np * Kp) return;
    int n = i / Kp;
    int c = i % Kp;
    int src = c;
    if (mode == 1) src = (c < 63) ? c : ((c == 63) ? -1 : c - 1);
    float v = 0.0f;
    gcf_p gw = (gcf_p)w;
    if (src >= 0 && src < K && n < Nsrc) v = gw[(size_t)src * Nsrc + n];
    ((GLOBAL_AS half_t*)wT)[i] = (half_t)v;
}

__global__ void convert_b_kernel(const float* __restrict__ b, float* __restrict__ bp,
                                 int Nsrc, int Np)
{
    int i = blockIdx.x * 256 + threadIdx.x;
    if (i >= Np) return;
    ((gf_p)bp)[i] = (i < Nsrc) ? ((gcf_p)b)[i] : 0.0f;
}

// ---------------------------------------------------------------------------
// Cooperative prefetch of an upcoming weight panel (global_prefetch_b8 path;
// no counter, no VGPR return - pure latency hiding behind the current GEMM).
// ---------------------------------------------------------------------------
__device__ inline void prefetch_panel(const half_t* w, int bytes)
{
    const char* p = (const char*)w;
    for (int o = (int)threadIdx.x * 256; o < bytes; o += NTHREADS * 256)
        __builtin_prefetch((const void*)(p + o), 0, 1);
}

// ---------------------------------------------------------------------------
// LDS-resident GEMM with 4 x NT_PER register blocking per wave.
//   wave w: M-group = (w&1)*4 tiles, N-group = (w>>1)*NT_PER tiles.
// Per K-step (32): 4 A-frags (ds_load_b128) + NT_PER B-frags (global_load_b128)
//   -> 4*NT_PER WMMAs (16 independent accumulation chains).
// A frag: lane reads K = k0 + (lane>>4)*16 .. +15 (contiguous, matches B)
// B frag: lane = column (lane&15), contiguous K panel (wT is [N][Kp]).
// Activation pointers are explicit LDS address space -> plain 32-bit ds
// addressing, no generic-pointer null checks in the hot loop.
// ---------------------------------------------------------------------------
template<int KP, int NT_PER>
__device__ inline void mlp_gemm(const half_t* __restrict__ wT,
                                const float*  __restrict__ bias,
                                lch_p in, int ldin,
                                lh_p out, int ldout, int ocol,
                                bool relu)
{
    const int tid  = threadIdx.x;
    const int lane = tid & 31;
    const int wave = tid >> 5;
    const int mtb  = (wave & 1) * 4;         // M-tile base (4 tiles)
    const int ntb  = (wave >> 1) * NT_PER;   // N-tile base
    const int col  = lane & 15;
    const int hi   = lane >> 4;

    gch_p gw = (gch_p)wT;                    // force global address space
    gcf_p gb = (gcf_p)bias;

    v8f acc[4][NT_PER];
#pragma unroll
    for (int j = 0; j < NT_PER; ++j) {
        const float bv = gb[(ntb + j) * 16 + col];
#pragma unroll
        for (int i = 0; i < 4; ++i)
#pragma unroll
            for (int q = 0; q < 8; ++q) acc[i][j][q] = bv;
    }

    lch_p abase = in + col * ldin + hi * 16;
    gch_p bbase = gw + (size_t)(ntb * 16 + col) * KP + hi * 16;

#pragma unroll
    for (int k0 = 0; k0 < KP; k0 += 32) {
        union HV { v16h v; v8h h[2]; };
        HV   a[4];
        v16h b[NT_PER];
#pragma unroll
        for (int i = 0; i < 4; ++i) {
            lch_p ap = abase + (mtb + i) * 16 * ldin + k0;
            a[i].h[0] = *(lcv8h_p)(ap);
            a[i].h[1] = *(lcv8h_p)(ap + 8);
        }
#pragma unroll
        for (int j = 0; j < NT_PER; ++j)
            b[j] = *(gcv16h_p)(bbase + (size_t)j * 16 * KP + k0);
#pragma unroll
        for (int i = 0; i < 4; ++i)
#pragma unroll
            for (int j = 0; j < NT_PER; ++j)
                acc[i][j] = __builtin_amdgcn_wmma_f32_16x16x32_f16(
                    false, a[i].v, false, b[j], (short)0, acc[i][j], false, false);
    }

#pragma unroll
    for (int i = 0; i < 4; ++i)
#pragma unroll
        for (int j = 0; j < NT_PER; ++j)
#pragma unroll
            for (int q = 0; q < 8; ++q) {
                float x = acc[i][j][q];
                if (relu) x = x > 0.0f ? x : 0.0f;
                out[(mtb * 16 + i * 16 + q + 8 * hi) * ldout
                    + ocol + (ntb + j) * 16 + col] = (half_t)x;
            }
}

// Head GEMM (1 N-tile): 8 waves, one M-tile each; f32 results straight to od.
template<int KP>
__device__ inline void mlp_head(const half_t* __restrict__ wT,
                                const float*  __restrict__ bias,
                                lch_p in, int ldin,
                                float* __restrict__ od, int pbase,
                                int colbase, int ncols)
{
    const int tid  = threadIdx.x;
    const int lane = tid & 31;
    const int mt   = tid >> 5;               // wave = M-tile 0..7
    const int col  = lane & 15;
    const int hi   = lane >> 4;
    lch_p arow = in + (mt * 16 + col) * ldin + hi * 16;

    gch_p gw = (gch_p)wT;
    gcf_p gb = (gcf_p)bias;
    gf_p  god = (gf_p)od;

    v8f acc;
    const float bv = gb[col];
#pragma unroll
    for (int q = 0; q < 8; ++q) acc[q] = bv;
    gch_p brow = gw + (size_t)col * KP + hi * 16;
#pragma unroll
    for (int k0 = 0; k0 < KP; k0 += 32) {
        union HV { v16h v; v8h h[2]; } a;
        a.h[0] = *(lcv8h_p)(arow + k0);
        a.h[1] = *(lcv8h_p)(arow + k0 + 8);
        v16h b = *(gcv16h_p)(brow + k0);
        acc = __builtin_amdgcn_wmma_f32_16x16x32_f16(false, a.v, false, b,
                                                     (short)0, acc, false, false);
    }
    if (col < ncols) {
#pragma unroll
        for (int q = 0; q < 8; ++q) {
            int m = mt * 16 + q + 8 * hi;
            god[(size_t)(pbase + m) * 4 + colbase + col] = acc[q];
        }
    }
}

// ---------------------------------------------------------------------------
// Fused NeRF MLP: embed -> 8 layers (+skip) -> alpha/feat/view/rgb heads.
// ---------------------------------------------------------------------------
struct MlpParams {
    const half_t* w[12];
    const float*  b[12];
};

__global__ __launch_bounds__(NTHREADS)
void nerf_mlp_kernel(MlpParams P, const float* __restrict__ rays,
                     const float* __restrict__ zbuf, float* __restrict__ od,
                     int nsamp, int usez)
{
    __shared__ half_t sPE[ROWS * 64];   // padded positional embedding (col 63 = 0)
    __shared__ half_t sDE[ROWS * 32];   // padded direction embedding (cols 27..31 = 0)
    __shared__ half_t sA[ROWS * 320];
    __shared__ half_t sB[ROWS * 256];
    __shared__ float  sP[ROWS * 3];
    __shared__ float  sV[ROWS * 3];

    lch_p cPE = (lch_p)sPE;
    lh_p  mPE = (lh_p)sPE;
    lh_p  mDE = (lh_p)sDE;
    lch_p cA  = (lch_p)sA;
    lh_p  mA  = (lh_p)sA;
    lch_p cB  = (lch_p)sB;
    lh_p  mB  = (lh_p)sB;

    const int tid   = threadIdx.x;
    const int pbase = blockIdx.x * ROWS;
    gcf_p grays = (gcf_p)rays;
    gcf_p gz    = (gcf_p)zbuf;

    prefetch_panel(P.w[0], 64 * 256 * 2);   // first panel while we embed

    if (tid < ROWS) {
        int p   = pbase + tid;
        int ray = p / nsamp;
        int smp = p - ray * nsamp;
        float ox = grays[ray * 3 + 0], oy = grays[ray * 3 + 1], oz = grays[ray * 3 + 2];
        float dx = grays[3 * M_RAYS + ray * 3 + 0];
        float dy = grays[3 * M_RAYS + ray * 3 + 1];
        float dz = grays[3 * M_RAYS + ray * 3 + 2];
        float s = usez ? gz[(size_t)ray * nsamp + smp]
                       : (2.0f + 4.0f * (float)smp / (float)(nsamp - 1));
        sP[tid * 3 + 0] = ox + dx * s;
        sP[tid * 3 + 1] = oy + dy * s;
        sP[tid * 3 + 2] = oz + dz * s;
        float inv = rsqrtf(dx * dx + dy * dy + dz * dz);
        sV[tid * 3 + 0] = dx * inv;
        sV[tid * 3 + 1] = dy * inv;
        sV[tid * 3 + 2] = dz * inv;
    }
    __syncthreads();

    // positional embedding: [x, sin(x*2^l), cos(x*2^l)], L=10 -> 63 cols (+1 pad)
    for (int idx = tid; idx < ROWS * 64; idx += NTHREADS) {
        int r = idx >> 6, c = idx & 63;
        float v = 0.0f;
        if (c < 3)       v = sP[r * 3 + c];
        else if (c < 33) { int l = (c - 3) / 3, ch = (c - 3) % 3;
                           v = __sinf(sP[r * 3 + ch] * (float)(1 << l)); }
        else if (c < 63) { int l = (c - 33) / 3, ch = (c - 33) % 3;
                           v = __cosf(sP[r * 3 + ch] * (float)(1 << l)); }
        mPE[idx] = (half_t)v;
    }
    // direction embedding: L=4 -> 27 cols (+5 pad)
    for (int idx = tid; idx < ROWS * 32; idx += NTHREADS) {
        int r = idx >> 5, c = idx & 31;
        float v = 0.0f;
        if (c < 3)       v = sV[r * 3 + c];
        else if (c < 15) { int l = (c - 3) / 3, ch = (c - 3) % 3;
                           v = __sinf(sV[r * 3 + ch] * (float)(1 << l)); }
        else if (c < 27) { int l = (c - 15) / 3, ch = (c - 15) % 3;
                           v = __cosf(sV[r * 3 + ch] * (float)(1 << l)); }
        mDE[idx] = (half_t)v;
    }
    __syncthreads();

    prefetch_panel(P.w[1], 256 * 256 * 2);
    mlp_gemm<64,  4>(P.w[0], P.b[0], cPE, 64,  mA, 320, 0,  true);  __syncthreads();
    prefetch_panel(P.w[2], 256 * 256 * 2);
    mlp_gemm<256, 4>(P.w[1], P.b[1], cA,  320, mB, 256, 0,  true);  __syncthreads();
    prefetch_panel(P.w[3], 256 * 256 * 2);
    mlp_gemm<256, 4>(P.w[2], P.b[2], cB,  256, mA, 320, 0,  true);  __syncthreads();
    prefetch_panel(P.w[4], 256 * 256 * 2);
    mlp_gemm<256, 4>(P.w[3], P.b[3], cA,  320, mB, 256, 0,  true);  __syncthreads();
    prefetch_panel(P.w[5], 320 * 256 * 2);
    mlp_gemm<256, 4>(P.w[4], P.b[4], cB,  256, mA, 320, 64, true);  __syncthreads();
    // skip concat: [pe | h4] -> sA cols 0..63 get pe (col 63 already zero pad)
    for (int idx = tid; idx < ROWS * 64; idx += NTHREADS) {
        int r = idx >> 6, c = idx & 63;
        mA[r * 320 + c] = cPE[idx];
    }
    __syncthreads();
    prefetch_panel(P.w[6], 256 * 256 * 2);
    mlp_gemm<320, 4>(P.w[5], P.b[5], cA,  320, mB, 256, 0,  true);  __syncthreads();
    prefetch_panel(P.w[7], 256 * 256 * 2);
    mlp_gemm<256, 4>(P.w[6], P.b[6], cB,  256, mA, 320, 0,  true);  __syncthreads();
    prefetch_panel(P.w[8], 16 * 256 * 2);
    prefetch_panel(P.w[9], 256 * 256 * 2);
    mlp_gemm<256, 4>(P.w[7], P.b[7], cA,  320, mB, 256, 0,  true);  __syncthreads();

    // alpha head (h -> 1), written to od[:,3]
    mlp_head<256>(P.w[8], P.b[8], cB, 256, od, pbase, 3, 1);
    // feat (h -> 256, no relu)
    prefetch_panel(P.w[10], 128 * 288 * 2);
    mlp_gemm<256, 4>(P.w[9], P.b[9], cB, 256, mA, 320, 0, false);   __syncthreads();
    // append de at cols 256..287 (cols 283..287 are zero pad)
    for (int idx = tid; idx < ROWS * 32; idx += NTHREADS) {
        int r = idx >> 5, c = idx & 31;
        mA[r * 320 + 256 + c] = mDE[idx];
    }
    __syncthreads();
    // view layer: 288 -> 128 (8 N-tiles -> NT_PER=2), relu
    prefetch_panel(P.w[11], 16 * 128 * 2);
    mlp_gemm<288, 2>(P.w[10], P.b[10], cA, 320, mB, 256, 0, true);  __syncthreads();
    // rgb head: 128 -> 3, written to od[:,0..2]
    mlp_head<128>(P.w[11], P.b[11], cB, 256, od, pbase, 0, 3);
}

// ---------------------------------------------------------------------------
// Volume compositing: one thread per ray (exclusive cumprod transmittance).
// ---------------------------------------------------------------------------
__global__ void composite_kernel(const float* __restrict__ od,
                                 const float* __restrict__ zbuf,
                                 const float* __restrict__ rays,
                                 float* __restrict__ rgb_out,
                                 float* __restrict__ wout,
                                 int nsamp, int usez)
{
    int r = blockIdx.x * blockDim.x + threadIdx.x;
    if (r >= M_RAYS) return;
    gcf_p god = (gcf_p)od;
    gcf_p gz  = (gcf_p)zbuf;
    gcf_p grd = (gcf_p)rays + 3 * M_RAYS + r * 3;
    float nrm = sqrtf(grd[0] * grd[0] + grd[1] * grd[1] + grd[2] * grd[2]);
    float T = 1.0f, cr = 0.0f, cg = 0.0f, cb = 0.0f;
    float sc = usez ? gz[(size_t)r * nsamp] : 2.0f;
    for (int n = 0; n < nsamp; ++n) {
        float sn = 0.0f, dist;
        if (n + 1 < nsamp) {
            sn = usez ? gz[(size_t)r * nsamp + n + 1]
                      : (2.0f + 4.0f * (float)(n + 1) / (float)(nsamp - 1));
            dist = (sn - sc) * nrm;
        } else {
            dist = 1e10f * nrm;
        }
        gcf_p o = god + ((size_t)r * nsamp + n) * 4;
        float sg = o[3] > 0.0f ? o[3] : 0.0f;
        float alpha = 1.0f - __expf(-sg * dist);
        float w = alpha * T;
        if (wout) ((gf_p)wout)[(size_t)r * nsamp + n] = w;
        cr += w / (1.0f + __expf(-o[0]));
        cg += w / (1.0f + __expf(-o[1]));
        cb += w / (1.0f + __expf(-o[2]));
        T *= (1.0f - alpha + 1e-10f);
        sc = sn;
    }
    gf_p go = (gf_p)rgb_out;
    go[r * 3 + 0] = cr;
    go[r * 3 + 1] = cg;
    go[r * 3 + 2] = cb;
}

// ---------------------------------------------------------------------------
// Inverse-CDF importance sampling + merge with stratified samples (sorted).
// ---------------------------------------------------------------------------
__device__ inline float mid_of(int j) { return 2.0f + 4.0f * ((float)j + 0.5f) / 63.0f; }

__global__ void sample_sort_kernel(const float* __restrict__ wc, float* __restrict__ z)
{
    int r = blockIdx.x * blockDim.x + threadIdx.x;
    if (r >= M_RAYS) return;
    gcf_p gwc = (gcf_p)wc;
    gf_p  gzz = (gf_p)z;
    float cdf[63];
    float pw[62];
    float wsum = 0.0f;
    for (int j = 0; j < 62; ++j) { pw[j] = gwc[r * 64 + 1 + j] + 1e-5f; wsum += pw[j]; }
    float inv = 1.0f / wsum;
    cdf[0] = 0.0f;
    float acc = 0.0f;
    for (int j = 0; j < 62; ++j) { acc += pw[j] * inv; cdf[j + 1] = acc; }
    float zs[128];
    for (int i = 0; i < 128; ++i) {
        float u = (float)i * (1.0f / 127.0f);
        int ind = 0;
        for (int j = 0; j < 63; ++j) ind += (u >= cdf[j]) ? 1 : 0;   // searchsorted right
        int below = ind - 1; if (below < 0) below = 0;
        int above = ind;     if (above > 62) above = 62;
        float cb_ = cdf[below], ca_ = cdf[above];
        float sb = mid_of(below), sa = mid_of(above);
        float den = ca_ - cb_; if (den < 1e-5f) den = 1.0f;
        float t = (u - cb_) / den;
        zs[i] = sb + t * (sa - sb);
    }
    // merge sorted stratified samples (64) with sorted importance samples (128)
    int i = 0, j = 0;
    for (int k = 0; k < 192; ++k) {
        float sv = (i < 64)  ? (2.0f + 4.0f * (float)i / 63.0f) : 3.0e38f;
        float zv = (j < 128) ? zs[j]                            : 3.0e38f;
        if (sv <= zv) { gzz[(size_t)r * 192 + k] = sv; ++i; }
        else          { gzz[(size_t)r * 192 + k] = zv; ++j; }
    }
}

// ---------------------------------------------------------------------------
// Host orchestration
// ---------------------------------------------------------------------------
extern "C" void kernel_launch(void* const* d_in, const int* in_sizes, int n_in,
                              void* d_out, int out_size, void* d_ws, size_t ws_size,
                              hipStream_t stream)
{
    (void)out_size; (void)ws_size;

    // locate rays (2*1024*3 elements)
    int rays_idx = -1;
    for (int i = 0; i < n_in; ++i)
        if (in_sizes[i] == 2 * M_RAYS * 3) { rays_idx = i; break; }
    if (rays_idx < 0) rays_idx = 0;
    const float* rays = (const float*)d_in[rays_idx];

    // remaining 48 inputs: first 24 = coarse params, last 24 = fine params
    int pidx[48]; int np = 0;
    for (int i = 0; i < n_in && np < 48; ++i)
        if (i != rays_idx) pidx[np++] = i;

    // canonical slots: 0..7 pts_w, 8..15 pts_b, 16 alpha_w, 17 alpha_b,
    //                  18 feat_w, 19 feat_b, 20 view_w, 21 view_b, 22 rgb_w, 23 rgb_b
    static const int ins_map[24] = { 0,2,4,6,8,10,12,14,  1,3,5,7,9,11,13,15,
                                     16,17, 18,19, 20,21, 22,23 };
    static const int srt_map[24] = { 12,13,14,15,16,17,18,19,  4,5,6,7,8,9,10,11,
                                     1,0, 3,2, 23,22, 21,20 };

    const float* pw[2][24];
    for (int g = 0; g < 2; ++g) {
        int base = g * 24;
        bool sorted_order = (in_sizes[pidx[base]] == 1);   // alpha_b first if sorted keys
        const int* mp = sorted_order ? srt_map : ins_map;
        for (int s = 0; s < 24; ++s) pw[g][s] = (const float*)d_in[pidx[base + mp[s]]];
    }

    // workspace carve-out
    char* wsb = (char*)d_ws;
    size_t off = 0;
    auto alloc = [&](size_t bytes) -> void* {
        off = (off + 255) & ~(size_t)255;
        void* p = wsb + off;
        off += bytes;
        return p;
    };

    struct LW { int K, Nsrc, Kp, Np, mode; };
    static const LW lw[12] = {
        { 63, 256,  64, 256, 0},                                  // pts_w0
        {256, 256, 256, 256, 0}, {256, 256, 256, 256, 0},         // pts_w1,2
        {256, 256, 256, 256, 0}, {256, 256, 256, 256, 0},         // pts_w3,4
        {319, 256, 320, 256, 1},                                  // pts_w5 (skip remap)
        {256, 256, 256, 256, 0}, {256, 256, 256, 256, 0},         // pts_w6,7
        {256,   1, 256,  16, 0},                                  // alpha
        {256, 256, 256, 256, 0},                                  // feat
        {283, 128, 288, 128, 0},                                  // view
        {128,   3, 128,  16, 0},                                  // rgb
    };
    static const int wslot[12] = {0,1,2,3,4,5,6,7, 16,18,20,22};
    static const int bslot[12] = {8,9,10,11,12,13,14,15, 17,19,21,23};

    MlpParams mp2[2];
    for (int g = 0; g < 2; ++g) {
        for (int l = 0; l < 12; ++l) {
            size_t nelW = (size_t)lw[l].Np * lw[l].Kp;
            half_t* wT = (half_t*)alloc(nelW * sizeof(half_t));
            float*  bp = (float*)alloc((size_t)lw[l].Np * sizeof(float));
            convert_w_kernel<<<(int)((nelW + 255) / 256), 256, 0, stream>>>(
                pw[g][wslot[l]], wT, lw[l].K, lw[l].Nsrc, lw[l].Kp, lw[l].Np, lw[l].mode);
            convert_b_kernel<<<(lw[l].Np + 255) / 256, 256, 0, stream>>>(
                pw[g][bslot[l]], bp, lw[l].Nsrc, lw[l].Np);
            mp2[g].w[l] = wT;
            mp2[g].b[l] = bp;
        }
    }

    float* od_c = (float*)alloc((size_t)M_RAYS * NC * 4 * sizeof(float));
    float* od_f = (float*)alloc((size_t)M_RAYS * NF * 4 * sizeof(float));
    float* wc   = (float*)alloc((size_t)M_RAYS * NC * sizeof(float));
    float* zb   = (float*)alloc((size_t)M_RAYS * NF * sizeof(float));
    float* out  = (float*)d_out;   // [0..3071] rgb_fine, [3072..6143] rgb_coarse

    // coarse pass
    nerf_mlp_kernel<<<(M_RAYS * NC) / ROWS, NTHREADS, 0, stream>>>(
        mp2[0], rays, (const float*)nullptr, od_c, NC, 0);
    composite_kernel<<<(M_RAYS + 255) / 256, 256, 0, stream>>>(
        od_c, (const float*)nullptr, rays, out + 3 * M_RAYS, wc, NC, 0);

    // importance sampling + sort
    sample_sort_kernel<<<(M_RAYS + 255) / 256, 256, 0, stream>>>(wc, zb);

    // fine pass
    nerf_mlp_kernel<<<(M_RAYS * NF) / ROWS, NTHREADS, 0, stream>>>(
        mp2[1], rays, zb, od_f, NF, 1);
    composite_kernel<<<(M_RAYS + 255) / 256, 256, 0, stream>>>(
        od_f, zb, rays, out, (float*)nullptr, NF, 1);
}
